// CausalMolSSM_89979564852031
// MI455X (gfx1250) — matrix operand order repaired and checked by reading, not compiled, not run on hardware
//
#include <hip/hip_runtime.h>
#include <hip/hip_bf16.h>
#include <math.h>

// ---------------------------------------------------------------------------
// Problem constants
// ---------------------------------------------------------------------------
#define SEQB   4
#define SEQL   4096
#define DMODEL 1024
#define FINNER 2048
#define MROWS  (SEQB * SEQL)      // 16384
#define NCH    8
#define CHUNK  (SEQL / NCH)       // 512

typedef __bf16 bf16_t;
typedef __attribute__((ext_vector_type(16))) __bf16 v16bf;
typedef __attribute__((ext_vector_type(8)))  float  v8f;
typedef __attribute__((ext_vector_type(4)))  int    v4i;

__device__ __forceinline__ float sigmoidf_(float v) { return 1.f / (1.f + __expf(-v)); }
__device__ __forceinline__ float siluf_(float v)    { return v * sigmoidf_(v); }
__device__ __forceinline__ float softplusf_(float v){ return (v > 20.f) ? v : log1pf(__expf(v)); }

// ---------------------------------------------------------------------------
// CDNA5 async global->LDS copy (ASYNCcnt-tracked), with sync fallback
// Signature (from compiler diagnostic): (int4 AS1* src, int4 AS3* dst, imm, imm)
// ---------------------------------------------------------------------------
#if __has_builtin(__builtin_amdgcn_global_load_async_to_lds_b128)
#define HAS_ASYNC_LDS 1
#else
#define HAS_ASYNC_LDS 0
#endif

#if HAS_ASYNC_LDS
typedef __attribute__((address_space(1))) v4i gas_v4i;
typedef __attribute__((address_space(3))) v4i las_v4i;
__device__ __forceinline__ void async_cp16(void* lds_dst, const void* gsrc) {
    __builtin_amdgcn_global_load_async_to_lds_b128(
        (gas_v4i*)gsrc, (las_v4i*)lds_dst, 0, 0);
}
__device__ __forceinline__ void wait_async0() {
#if __has_builtin(__builtin_amdgcn_s_wait_asynccnt)
    __builtin_amdgcn_s_wait_asynccnt(0);
#else
    asm volatile("s_wait_asynccnt 0x0" ::: "memory");
#endif
}
#endif

// ---------------------------------------------------------------------------
// f32 -> bf16 conversion (grid-stride)
// ---------------------------------------------------------------------------
__global__ void cvt_bf16_k(const float* __restrict__ in, bf16_t* __restrict__ out, long n) {
    long i = (long)blockIdx.x * blockDim.x + threadIdx.x;
    long stride = (long)gridDim.x * blockDim.x;
    for (; i < n; i += stride) out[i] = (bf16_t)in[i];
}

// ---------------------------------------------------------------------------
// bf16 WMMA GEMM:  C[M,N] = A[M,K] * Bw[N,K]^T   (row-major, K contiguous)
// Block tile 128x128, K-step 32, double-buffered LDS, async global->LDS.
// 8 waves, each wave owns a 32x64 output tile (2x4 WMMA accumulators).
// Fused epilogues:
//   EPI 0: out0[m,n] = acc                           (out_proj -> d_out)
//   EPI 1: n<F: out0=acc (x1) ; n>=F: out1=silu(acc) (in_proj split+gate)
//   EPI 2: out0 = softplus(acc + bias[n])            (dt)
//   EPI 3: out0 = (float)auxbf[m,n] * sigmoid(acc)   (u = xc * B_t)
//   EPI 4: out0 = tanh(acc)                          (C_t)
// ---------------------------------------------------------------------------
#define BM 128
#define BN 128
#define BK 32
#define LDSW 40   // halfs per LDS row (padded: 80B rows, 16B aligned)

__device__ __forceinline__ void wmma_tile(
    const bf16_t* __restrict__ sa, const bf16_t* __restrict__ sb,
    int wm, int wn, int col, int rsel, v8f (&acc)[2][4])
{
    union Frag { v16bf v; uint4 q[2]; } a[2], b[4];
    // A fragment: lanes 0-15 -> K 0-7 & 16-23 ; lanes 16-31 -> K 8-15 & 24-31
    #pragma unroll
    for (int mi = 0; mi < 2; mi++) {
        int r = wm * 32 + mi * 16 + col;
        a[mi].q[0] = *(const uint4*)(&sa[r * LDSW + rsel * 8]);
        a[mi].q[1] = *(const uint4*)(&sa[r * LDSW + 16 + rsel * 8]);
    }
    // B fragment: lanes 0-15 -> K 0-7 & 16-23 ; lanes 16-31 -> K 8-15 & 24-31
    #pragma unroll
    for (int ni = 0; ni < 4; ni++) {
        int r = wn * 64 + ni * 16 + col;
        b[ni].q[0] = *(const uint4*)(&sb[r * LDSW + rsel * 8]);
        b[ni].q[1] = *(const uint4*)(&sb[r * LDSW + 16 + rsel * 8]);
    }
    #pragma unroll
    for (int mi = 0; mi < 2; mi++)
        #pragma unroll
        for (int ni = 0; ni < 4; ni++)
            acc[mi][ni] = __builtin_amdgcn_wmma_f32_16x16x32_bf16(
                false, a[mi].v, false, b[ni].v,
                (short)0, acc[mi][ni], false, false);
}

template<int EPI>
__global__ __launch_bounds__(256) void gemm_bf16_k(
    const bf16_t* __restrict__ A, const bf16_t* __restrict__ Bw,
    float* __restrict__ out0, float* __restrict__ out1,
    const float* __restrict__ bias, const bf16_t* __restrict__ auxbf,
    int M, int N, int K)
{
    __shared__ __align__(16) bf16_t sA[2][BM * LDSW];
    __shared__ __align__(16) bf16_t sB[2][BN * LDSW];

    const int tid  = threadIdx.x;
    const int lane = tid & 31;
    const int w    = tid >> 5;      // wave 0..7
    const int wm   = w >> 1;        // 0..3 along M
    const int wn   = w & 1;         // 0..1 along N
    const int m0   = blockIdx.y * BM;
    const int n0   = blockIdx.x * BN;
    const int col  = lane & 15;
    const int rsel = lane >> 4;

    // Branch-free staging: thread owns chunks (row, c8) and (row+64, c8); 16B each
    const int srow = tid >> 2;              // 0..63
    const int sc8  = (tid & 3) * 8;         // half offset 0/8/16/24
    const bf16_t* gA = A  + (size_t)(m0 + srow) * K + sc8;
    const bf16_t* gB = Bw + (size_t)(n0 + srow) * K + sc8;
    const size_t gRowJump = (size_t)64 * K;
    const int lOff = srow * LDSW + sc8;
    const int lJump = 64 * LDSW;

    v8f acc[2][4];
    #pragma unroll
    for (int i = 0; i < 2; i++)
        #pragma unroll
        for (int j = 0; j < 4; j++)
            #pragma unroll
            for (int e = 0; e < 8; e++) acc[i][j][e] = 0.f;

    const int nk = K / BK;

#if HAS_ASYNC_LDS
    // prologue: stage tile 0 into buffer 0 via async copies
    async_cp16(&sA[0][lOff],         gA);
    async_cp16(&sA[0][lOff + lJump], gA + gRowJump);
    async_cp16(&sB[0][lOff],         gB);
    async_cp16(&sB[0][lOff + lJump], gB + gRowJump);
    wait_async0();
    __syncthreads();

    for (int kt = 0; kt < nk; kt++) {
        const int cur = kt & 1;
        if (kt + 1 < nk) {
            const int k1 = (kt + 1) * BK;
            async_cp16(&sA[cur ^ 1][lOff],         gA + k1);
            async_cp16(&sA[cur ^ 1][lOff + lJump], gA + gRowJump + k1);
            async_cp16(&sB[cur ^ 1][lOff],         gB + k1);
            async_cp16(&sB[cur ^ 1][lOff + lJump], gB + gRowJump + k1);
        }
        wmma_tile(sA[cur], sB[cur], wm, wn, col, rsel, acc);
        wait_async0();
        __syncthreads();
    }
#else
    // prologue: register-staged copy of tile 0
    {
        uint4 r0 = *(const uint4*)(gA);
        uint4 r1 = *(const uint4*)(gA + gRowJump);
        uint4 r2 = *(const uint4*)(gB);
        uint4 r3 = *(const uint4*)(gB + gRowJump);
        *(uint4*)(&sA[0][lOff])         = r0;
        *(uint4*)(&sA[0][lOff + lJump]) = r1;
        *(uint4*)(&sB[0][lOff])         = r2;
        *(uint4*)(&sB[0][lOff + lJump]) = r3;
    }
    __syncthreads();

    for (int kt = 0; kt < nk; kt++) {
        const int cur = kt & 1;
        uint4 r0, r1, r2, r3;
        const bool pf = (kt + 1 < nk);
        if (pf) {   // issue global loads for next tile BEFORE compute
            const int k1 = (kt + 1) * BK;
            r0 = *(const uint4*)(gA + k1);
            r1 = *(const uint4*)(gA + gRowJump + k1);
            r2 = *(const uint4*)(gB + k1);
            r3 = *(const uint4*)(gB + gRowJump + k1);
        }
        wmma_tile(sA[cur], sB[cur], wm, wn, col, rsel, acc);
        if (pf) {
            *(uint4*)(&sA[cur ^ 1][lOff])         = r0;
            *(uint4*)(&sA[cur ^ 1][lOff + lJump]) = r1;
            *(uint4*)(&sB[cur ^ 1][lOff])         = r2;
            *(uint4*)(&sB[cur ^ 1][lOff + lJump]) = r3;
        }
        __syncthreads();
    }
#endif

    // Epilogue: lane holds column (lane&15); rows (lane>>4)*8 + i per 16x16 tile
    #pragma unroll
    for (int mi = 0; mi < 2; mi++) {
        #pragma unroll
        for (int ni = 0; ni < 4; ni++) {
            int gn = n0 + wn * 64 + ni * 16 + col;
            #pragma unroll
            for (int i = 0; i < 8; i++) {
                int gm = m0 + wm * 32 + mi * 16 + rsel * 8 + i;
                float v = acc[mi][ni][i];
                if constexpr (EPI == 0) {
                    out0[(size_t)gm * N + gn] = v;
                } else if constexpr (EPI == 1) {
                    if (gn < FINNER) out0[(size_t)gm * FINNER + gn] = v;
                    else             out1[(size_t)gm * FINNER + (gn - FINNER)] = siluf_(v);
                } else if constexpr (EPI == 2) {
                    out0[(size_t)gm * N + gn] = softplusf_(v + bias[gn]);
                } else if constexpr (EPI == 3) {
                    out0[(size_t)gm * N + gn] =
                        (float)auxbf[(size_t)gm * N + gn] * sigmoidf_(v);
                } else {
                    out0[(size_t)gm * N + gn] = tanhf(v);
                }
            }
        }
    }
}

// ---------------------------------------------------------------------------
// Depthwise causal conv1d (k=4) + bias + SiLU -> bf16 xc
// ---------------------------------------------------------------------------
__global__ __launch_bounds__(256) void dwconv_silu_k(
    const float* __restrict__ xin, const float* __restrict__ Wconv,
    const float* __restrict__ bconv, bf16_t* __restrict__ xcbf)
{
    int tid = blockIdx.x * 256 + threadIdx.x;       // b*F + f
    if (tid >= SEQB * FINNER) return;
    int f = tid & (FINNER - 1);
    int b = tid / FINNER;
    float w0 = Wconv[f * 4 + 0], w1 = Wconv[f * 4 + 1];
    float w2 = Wconv[f * 4 + 2], w3 = Wconv[f * 4 + 3];
    float bia = bconv[f];
    const float* p = xin + (size_t)b * SEQL * FINNER + f;
    bf16_t*      q = xcbf + (size_t)b * SEQL * FINNER + f;
    float a0 = 0.f, a1 = 0.f, a2 = 0.f;
    for (int l = 0; l < SEQL; l++) {
        float a3 = p[(size_t)l * FINNER];
        float v = w0 * a0 + w1 * a1 + w2 * a2 + w3 * a3 + bia;
        q[(size_t)l * FINNER] = (bf16_t)siluf_(v);
        a0 = a1; a1 = a2; a2 = a3;
    }
}

// ---------------------------------------------------------------------------
// Chunked linear scan:  h_t = alpha_t * h_{t-1} + beta_t
//   alpha_t = exp(clip(A_f*dt_t, -10, -1e-4)),  beta_t = dt_t*0.5*(u_{t-1}+u_t)
// ---------------------------------------------------------------------------
__global__ __launch_bounds__(256) void scan_p1_k(
    const float* __restrict__ dt, const float* __restrict__ u,
    const float* __restrict__ A_log,
    float* __restrict__ Pc, float* __restrict__ He)
{
    int tid = blockIdx.x * 256 + threadIdx.x;       // (b*NCH + c)*F + f
    if (tid >= SEQB * NCH * FINNER) return;
    int f = tid & (FINNER - 1);
    int c = (tid / FINNER) & (NCH - 1);
    int b = tid / (FINNER * NCH);
    float Af = -__expf(A_log[f]);
    size_t base = ((size_t)b * SEQL + (size_t)c * CHUNK) * FINNER + f;
    float up = (c > 0) ? u[base - FINNER] : 0.f;
    float P = 1.f, h = 0.f;
    for (int i = 0; i < CHUNK; i++) {
        size_t idx = base + (size_t)i * FINNER;
        float d = dt[idx], uu = u[idx];
        float al = __expf(fminf(fmaxf(Af * d, -10.f), -1e-4f));
        h = al * h + d * 0.5f * (up + uu);
        P *= al;
        up = uu;
    }
    size_t o = ((size_t)b * NCH + c) * FINNER + f;
    Pc[o] = P; He[o] = h;
}

__global__ __launch_bounds__(256) void scan_p2_k(
    const float* __restrict__ Pc, const float* __restrict__ He,
    float* __restrict__ hin)
{
    int tid = blockIdx.x * 256 + threadIdx.x;       // b*F + f
    if (tid >= SEQB * FINNER) return;
    int f = tid & (FINNER - 1);
    int b = tid / FINNER;
    float h = 0.f;
    for (int c = 0; c < NCH; c++) {
        size_t o = ((size_t)b * NCH + c) * FINNER + f;
        hin[o] = h;
        h = Pc[o] * h + He[o];
    }
}

__global__ __launch_bounds__(256) void scan_p3_k(
    const float* __restrict__ dt, const float* __restrict__ u,
    const float* __restrict__ A_log, const float* __restrict__ hin,
    const float* __restrict__ Ct, const float* __restrict__ sz,
    bf16_t* __restrict__ ybf)
{
    int tid = blockIdx.x * 256 + threadIdx.x;
    if (tid >= SEQB * NCH * FINNER) return;
    int f = tid & (FINNER - 1);
    int c = (tid / FINNER) & (NCH - 1);
    int b = tid / (FINNER * NCH);
    float Af = -__expf(A_log[f]);
    size_t base = ((size_t)b * SEQL + (size_t)c * CHUNK) * FINNER + f;
    float up = (c > 0) ? u[base - FINNER] : 0.f;
    float h = hin[((size_t)b * NCH + c) * FINNER + f];
    for (int i = 0; i < CHUNK; i++) {
        size_t idx = base + (size_t)i * FINNER;
        float d = dt[idx], uu = u[idx];
        float al = __expf(fminf(fmaxf(Af * d, -10.f), -1e-4f));
        h = al * h + d * 0.5f * (up + uu);
        up = uu;
        ybf[idx] = (bf16_t)(h * Ct[idx] * sz[idx]);
    }
}

// ---------------------------------------------------------------------------
// Host-side launcher
// ---------------------------------------------------------------------------
extern "C" void kernel_launch(void* const* d_in, const int* in_sizes, int n_in,
                              void* d_out, int out_size, void* d_ws, size_t ws_size,
                              hipStream_t stream)
{
    const float* x     = (const float*)d_in[0];
    const float* Wi    = (const float*)d_in[1];
    const float* Wconv = (const float*)d_in[2];
    const float* bconv = (const float*)d_in[3];
    const float* Wdt   = (const float*)d_in[4];
    const float* bdt   = (const float*)d_in[5];
    const float* WB    = (const float*)d_in[6];
    const float* WC    = (const float*)d_in[7];
    const float* Wo    = (const float*)d_in[8];
    const float* A_log = (const float*)d_in[9];
    float* out = (float*)d_out;

    char* ws = (char*)d_ws;
    size_t off = 0;
    auto alloc = [&](size_t bytes) -> void* {
        void* p = ws + off;
        off = (off + bytes + 255) & ~(size_t)255;
        return p;
    };

    const size_t MF = (size_t)MROWS * FINNER;
    bf16_t* xbf   = (bf16_t*)alloc((size_t)MROWS * DMODEL * 2);
    bf16_t* wi_b  = (bf16_t*)alloc((size_t)(2 * FINNER) * DMODEL * 2);
    bf16_t* wdt_b = (bf16_t*)alloc((size_t)FINNER * FINNER * 2);
    bf16_t* wb_b  = (bf16_t*)alloc((size_t)FINNER * FINNER * 2);
    bf16_t* wc_b  = (bf16_t*)alloc((size_t)FINNER * FINNER * 2);
    bf16_t* wo_b  = (bf16_t*)alloc((size_t)DMODEL * FINNER * 2);
    float*  x1    = (float*)alloc(MF * 4);
    float*  sz    = (float*)alloc(MF * 4);
    bf16_t* xcbf  = (bf16_t*)alloc(MF * 2);
    float*  dtb   = (float*)alloc(MF * 4);
    float*  ub    = (float*)alloc(MF * 4);
    float*  ctb   = (float*)alloc(MF * 4);
    bf16_t* ybf   = (bf16_t*)alloc(MF * 2);
    float*  Pc    = (float*)alloc((size_t)SEQB * NCH * FINNER * 4);
    float*  He    = (float*)alloc((size_t)SEQB * NCH * FINNER * 4);
    float*  hinb  = (float*)alloc((size_t)SEQB * NCH * FINNER * 4);

    cvt_bf16_k<<<2048, 256, 0, stream>>>(x,   xbf,   (long)MROWS * DMODEL);
    cvt_bf16_k<<<1024, 256, 0, stream>>>(Wi,  wi_b,  (long)2 * FINNER * DMODEL);
    cvt_bf16_k<<<1024, 256, 0, stream>>>(Wdt, wdt_b, (long)FINNER * FINNER);
    cvt_bf16_k<<<1024, 256, 0, stream>>>(WB,  wb_b,  (long)FINNER * FINNER);
    cvt_bf16_k<<<1024, 256, 0, stream>>>(WC,  wc_b,  (long)FINNER * FINNER);
    cvt_bf16_k<<<1024, 256, 0, stream>>>(Wo,  wo_b,  (long)DMODEL * FINNER);

    {   // in_proj: xz = x Wi^T ; split -> x1, silu(z)
        dim3 g((2 * FINNER) / BN, MROWS / BM);
        gemm_bf16_k<1><<<g, 256, 0, stream>>>(xbf, wi_b, x1, sz, nullptr, nullptr,
                                              MROWS, 2 * FINNER, DMODEL);
    }
    dwconv_silu_k<<<(SEQB * FINNER) / 256, 256, 0, stream>>>(x1, Wconv, bconv, xcbf);

    {   // dt / u / C_t projections, fused activations
        dim3 g(FINNER / BN, MROWS / BM);
        gemm_bf16_k<2><<<g, 256, 0, stream>>>(xcbf, wdt_b, dtb, nullptr, bdt, nullptr,
                                              MROWS, FINNER, FINNER);
        gemm_bf16_k<3><<<g, 256, 0, stream>>>(xcbf, wb_b, ub, nullptr, nullptr, xcbf,
                                              MROWS, FINNER, FINNER);
        gemm_bf16_k<4><<<g, 256, 0, stream>>>(xcbf, wc_b, ctb, nullptr, nullptr, nullptr,
                                              MROWS, FINNER, FINNER);
    }

    scan_p1_k<<<(SEQB * NCH * FINNER) / 256, 256, 0, stream>>>(dtb, ub, A_log, Pc, He);
    scan_p2_k<<<(SEQB * FINNER) / 256, 256, 0, stream>>>(Pc, He, hinb);
    scan_p3_k<<<(SEQB * NCH * FINNER) / 256, 256, 0, stream>>>(dtb, ub, A_log, hinb,
                                                               ctb, sz, ybf);

    {   // out_proj
        dim3 g(DMODEL / BN, MROWS / BM);
        gemm_bf16_k<0><<<g, 256, 0, stream>>>(ybf, wo_b, out, nullptr, nullptr, nullptr,
                                              MROWS, DMODEL, FINNER);
    }
}
